// HistoryAttention_56762287784353
// MI455X (gfx1250) — matrix operand order repaired and checked by reading, not compiled, not run on hardware
//
#include <hip/hip_runtime.h>
#include <hip/hip_bf16.h>

// ---------------------------------------------------------------------------
// HistoryAttention for MI455X (gfx1250, wave32, WMMA).
//
// Math restructuring: S = (Q@Wk) . hist  (bk softmax-invariant),
//                     attended = (softmax@hist) @ Wv^T + bv.
// => 4 dense [16384x1024]x[1024x1024] bf16 WMMA GEMMs (~34 GF each) + a tiny
//    per-position attention core (H=32). history read exactly once (256 MB).
//
// GEMM: 256-thread WG computes a 256x128 tile. B (weight) 128x32 k-slab is
// staged in LDS once per WG (double-buffered), each wave owns a 32x128 strip
// (2 M-tiles x 8 N-tiles = 16 v_wmma per k-step) so each LDS B fragment
// feeds two WMMAs.
// ---------------------------------------------------------------------------

#define Bsz 8
#define Tsz 2048
#define Dm  1024
#define Hh  32
#define HSTRIDE 1032  // attn LDS row stride (elems); 2064B = 129*16B
#define BSTRIDE 40    // gemm LDS col stride (elems); 80B = 20 dwords -> no bank conflicts

typedef __attribute__((ext_vector_type(16))) __bf16 v16bf;
typedef __attribute__((ext_vector_type(8)))  __bf16 v8bf;
typedef __attribute__((ext_vector_type(8)))  float  v8f;

// ---------------------------------------------------------------------------
// Weight conversion: Wq/Wv/Wo as-is (row n contiguous in k == B-fragment
// layout for X@W^T), Wk transposed (so Qk=Q@Wk also gets contiguous-K rows).
// ---------------------------------------------------------------------------
__global__ __launch_bounds__(256) void convert_weights_kernel(
    const float* __restrict__ Wq, const float* __restrict__ Wk,
    const float* __restrict__ Wv, const float* __restrict__ Wo,
    __bf16* __restrict__ wq, __bf16* __restrict__ wkT,
    __bf16* __restrict__ wv, __bf16* __restrict__ wo)
{
    int i = blockIdx.x * 256 + threadIdx.x;     // i < Dm*Dm (grid = 4096)
    wq[i] = (__bf16)Wq[i];
    wv[i] = (__bf16)Wv[i];
    wo[i] = (__bf16)Wo[i];
    int n = i >> 10, k = i & (Dm - 1);
    wkT[i] = (__bf16)Wk[k * Dm + n];            // wkT[n][k] = Wk[k][n]
}

__global__ __launch_bounds__(256) void convert_x_kernel(
    const float* __restrict__ x, __bf16* __restrict__ xb)
{
    size_t i = ((size_t)blockIdx.x * 256 + threadIdx.x) * 4;
    float4 v = *(const float4*)(x + i);
    __bf16* o = xb + i;
    o[0] = (__bf16)v.x; o[1] = (__bf16)v.y;
    o[2] = (__bf16)v.z; o[3] = (__bf16)v.w;
}

// ---------------------------------------------------------------------------
// bf16 WMMA GEMM: C[M,1024] = A[M,1024] @ Bw^T + bias (+resid), Bw is [N][K]
// row-major. Fragment layouts used here (contiguous k, same order in A and B):
//   A 16x32: lane<16 holds K0..15 of row lane&15; lane>=16 K16..31 (contig per lane)
//   B 32x16: lane&15 = col, lanes<16 K0..15, lanes>=16 K16..31 (contig per lane)
//   C 16x16: c[r] -> M = r + 8*(lane>=16), N = lane&15
// ---------------------------------------------------------------------------
template<bool OUT_BF16, bool RESID>
__global__ __launch_bounds__(256) void gemm_kernel(
    const __bf16* __restrict__ A, const __bf16* __restrict__ Bw,
    const float* __restrict__ bias, const float* __restrict__ resid,
    void* __restrict__ out)
{
    __shared__ __bf16 ldsB[2][128 * BSTRIDE];   // 2 x 10 KB, double-buffered

    const int tid   = threadIdx.x;
    const int lane  = tid & 31;
    const int wave  = tid >> 5;
    const int tileN = blockIdx.x * 128;
    const int tileM = blockIdx.y * 256 + wave * 32;
    const int l15   = lane & 15;
    const int koff  = (lane < 16) ? 0 : 16;   // A half-wave K offset
    const int kgrp  = (lane < 16) ? 0 : 16;   // B half-wave K offset

    // B staging: each thread owns one 32B chunk of the 128x32 k-slab
    const int scol  = tid >> 1;               // 0..127 (N within tile)
    const int shalf = (tid & 1) * 16;         // k offset 0 / 16
    const __bf16* bsrc = Bw + (size_t)(tileN + scol) * Dm + shalf;
    const int sdst = scol * BSTRIDE + shalf;

    v8f acc[2][8] = {};
    const __bf16* aptr0 = A + (size_t)(tileM + l15) * Dm;
    const __bf16* aptr1 = aptr0 + (size_t)16 * Dm;

    // prologue: stage k0 = 0 slab into buffer 0
    v16bf pre = *(const v16bf*)bsrc;
    *(v16bf*)(&ldsB[0][0] + sdst) = pre;
    __syncthreads();

    for (int k0 = 0; k0 < Dm; k0 += 32) {
        const int  buf  = (k0 >> 5) & 1;
        const bool more = (k0 + 32) < Dm;
        if (more) pre = *(const v16bf*)(bsrc + k0 + 32);   // global prefetch

        // A fragments for the wave's two M-tiles (direct from global)
        v16bf af0, af1;
        {
            v8bf a0 = *(const v8bf*)(aptr0 + k0 + koff);
            v8bf a1 = *(const v8bf*)(aptr0 + k0 + koff + 8);
            v8bf c0 = *(const v8bf*)(aptr1 + k0 + koff);
            v8bf c1 = *(const v8bf*)(aptr1 + k0 + koff + 8);
#pragma unroll
            for (int e = 0; e < 8; ++e) {
                af0[e] = a0[e]; af0[8 + e] = a1[e];
                af1[e] = c0[e]; af1[8 + e] = c1[e];
            }
        }
#pragma unroll
        for (int j = 0; j < 8; ++j) {
            const __bf16* bp = &ldsB[buf][0] + (j * 16 + l15) * BSTRIDE + kgrp;
            v16bf bfr = *(const v16bf*)bp;                 // 32B LDS read
            acc[0][j] = __builtin_amdgcn_wmma_f32_16x16x32_bf16(
                false, af0, false, bfr, (short)0, acc[0][j], false, false);
            acc[1][j] = __builtin_amdgcn_wmma_f32_16x16x32_bf16(
                false, af1, false, bfr, (short)0, acc[1][j], false, false);
        }
        if (more) *(v16bf*)(&ldsB[buf ^ 1][0] + sdst) = pre;
        __syncthreads();
    }

    const int rowOutBase = tileM + ((lane >> 4) << 3);
#pragma unroll
    for (int mt = 0; mt < 2; ++mt) {
#pragma unroll
        for (int j = 0; j < 8; ++j) {
            const int col = tileN + j * 16 + l15;
            const float bb = bias ? bias[col] : 0.0f;
#pragma unroll
            for (int r = 0; r < 8; ++r) {
                const size_t idx = (size_t)(rowOutBase + mt * 16 + r) * Dm + col;
                float v = acc[mt][j][r] + bb;
                if (RESID) v += resid[idx];
                if (OUT_BF16) ((__bf16*)out)[idx] = (__bf16)v;
                else          ((float*)out)[idx]  = v;
            }
        }
    }
}

// ---------------------------------------------------------------------------
// Attention core: one workgroup per position t.
//  1) stage hist[t] (32x1024 f32 -> bf16) into LDS (padded rows)
//  2) thread(b=wave, h=lane): S[b][h] = Qk[b,t,:].hist[t,h,:]/32 + log(decay)
//  3) wave32 shfl softmax over h; attn -> LDS
//  4) P[b,t,d] = sum_h attn[b,h]*hist[t,h,d]  (tiny, VALU from LDS)
// ---------------------------------------------------------------------------
__global__ __launch_bounds__(256) void attn_kernel(
    const __bf16* __restrict__ Qk, const float* __restrict__ hist,
    const float* __restrict__ decay, __bf16* __restrict__ P)
{
    extern __shared__ char smem[];
    __bf16* histb = (__bf16*)smem;                          // [32][HSTRIDE]
    float*  attnw = (float*)(smem + Hh * HSTRIDE * 2);      // [8][32]

    const int t   = blockIdx.x;
    const int tid = threadIdx.x;

    // stage hist[t] -> bf16 LDS (each thread: 16 chunks of 8 elems)
    const float* hsrc = hist + (size_t)t * (Hh * Dm);
    for (int c = tid; c < (Hh * Dm / 8); c += 256) {
        int h = c >> 7;
        int d = (c & 127) * 8;
        const float4* p4 = (const float4*)(hsrc + h * Dm + d);
        float4 f0 = p4[0], f1 = p4[1];
        v8bf o;
        o[0] = (__bf16)f0.x; o[1] = (__bf16)f0.y;
        o[2] = (__bf16)f0.z; o[3] = (__bf16)f0.w;
        o[4] = (__bf16)f1.x; o[5] = (__bf16)f1.y;
        o[6] = (__bf16)f1.z; o[7] = (__bf16)f1.w;
        *(v8bf*)(histb + h * HSTRIDE + d) = o;
    }
    __syncthreads();

    const int b = tid >> 5;   // wave id
    const int h = tid & 31;   // lane id

    // scores
    const __bf16* qrow = Qk + ((size_t)b * Tsz + t) * Dm;   // wave-uniform
    float s = 0.0f;
    for (int d = 0; d < Dm; d += 8) {
        v8bf q8 = *(const v8bf*)(qrow + d);
        v8bf h8 = *(const v8bf*)(histb + h * HSTRIDE + d);
#pragma unroll
        for (int e = 0; e < 8; ++e) s += (float)q8[e] * (float)h8[e];
    }
    s *= 0.03125f;                                  // 1/sqrt(1024)
    s += logf(decay[Hh - 1 - h] + 1e-10f);          // flipped decay bias

    // wave32 softmax over h
    float m = s;
#pragma unroll
    for (int off = 16; off; off >>= 1) m = fmaxf(m, __shfl_xor(m, off));
    float e = expf(s - m);
    float sum = e;
#pragma unroll
    for (int off = 16; off; off >>= 1) sum += __shfl_xor(sum, off);
    attnw[b * 32 + h] = e / sum;
    __syncthreads();

    // P = attn @ hist[t]
    for (int d = tid; d < Dm; d += 256) {
        float acc[8] = {};
#pragma unroll 8
        for (int hh = 0; hh < Hh; ++hh) {
            float hv = (float)histb[hh * HSTRIDE + d];
#pragma unroll
            for (int bb = 0; bb < 8; ++bb)
                acc[bb] += attnw[bb * 32 + hh] * hv;
        }
#pragma unroll
        for (int bb = 0; bb < 8; ++bb)
            P[((size_t)bb * Tsz + t) * Dm + d] = (__bf16)acc[bb];
    }
}

// ---------------------------------------------------------------------------
extern "C" void kernel_launch(void* const* d_in, const int* in_sizes, int n_in,
                              void* d_out, int out_size, void* d_ws, size_t ws_size,
                              hipStream_t stream)
{
    (void)in_sizes; (void)n_in; (void)out_size; (void)ws_size;

    const float* x     = (const float*)d_in[0];
    const float* hist  = (const float*)d_in[1];
    const float* Wq    = (const float*)d_in[2];
    const float* bq    = (const float*)d_in[3];
    const float* Wk    = (const float*)d_in[4];
    /* bk (d_in[5]) is softmax-invariant -> dropped */
    const float* Wv    = (const float*)d_in[6];
    const float* bv    = (const float*)d_in[7];
    const float* Wo    = (const float*)d_in[8];
    const float* bo    = (const float*)d_in[9];
    const float* decay = (const float*)d_in[10];

    const size_t DD = (size_t)Dm * Dm;            // 1M elems
    const size_t MT = (size_t)Bsz * Tsz * Dm;     // 16.7M elems

    __bf16* wq  = (__bf16*)d_ws;
    __bf16* wkT = wq  + DD;
    __bf16* wv  = wkT + DD;
    __bf16* wo  = wv  + DD;
    __bf16* xb  = wo  + DD;       // x  bf16   (32 MB)
    __bf16* qb  = xb  + MT;       // Q  bf16   (32 MB)
    __bf16* qkb = qb  + MT;       // Qk bf16   (32 MB)
    __bf16* pb  = qb;             // reuse Q buffer for P
    __bf16* yb  = xb;             // reuse x_bf buffer for y

    convert_weights_kernel<<<DD / 256, 256, 0, stream>>>(Wq, Wk, Wv, Wo,
                                                         wq, wkT, wv, wo);
    convert_x_kernel<<<MT / (4 * 256), 256, 0, stream>>>(x, xb);

    dim3 g(Dm / 128, (Bsz * Tsz) / 256);   // (8, 64)

    // Q = x @ Wq^T + bq
    gemm_kernel<true, false><<<g, 256, 0, stream>>>(xb, wq, bq, nullptr, qb);
    // Qk = Q @ Wk
    gemm_kernel<true, false><<<g, 256, 0, stream>>>(qb, wkT, nullptr, nullptr, qkb);

    // attention core
    int smem = Hh * HSTRIDE * 2 + Bsz * Hh * (int)sizeof(float);
    (void)hipFuncSetAttribute(reinterpret_cast<const void*>(attn_kernel),
                              hipFuncAttributeMaxDynamicSharedMemorySize, smem);
    attn_kernel<<<Tsz, 256, smem, stream>>>(qkb, hist, decay, pb);

    // y = P @ Wv^T + bv + x
    gemm_kernel<true, true><<<g, 256, 0, stream>>>(pb, wv, bv, x, yb);
    // out = y @ Wo^T + bo
    gemm_kernel<false, false><<<g, 256, 0, stream>>>(yb, wo, bo, nullptr, d_out);
}
